// lstm_2345052144277
// MI455X (gfx1250) — matrix operand order repaired and checked by reading, not compiled, not run on hardware
//
#include <hip/hip_runtime.h>

// ---------------------------------------------------------------------------
// 3-layer relu-LSTM + dense head for MI455X (gfx1250, wave32, WMMA bf16).
// Strategy: fused [x_t ; h] @ [W ; U] per-step GEMM with weights resident in
// LDS (CDNA5: 320KB/WGP), one workgroup per 16-row batch tile (recurrence is
// independent across batch), bf16 WMMA 16x16x32 with f32 accumulation.
// ---------------------------------------------------------------------------

typedef __attribute__((ext_vector_type(16))) __bf16 v16bf;
typedef __attribute__((ext_vector_type(8)))  float  v8f;

union FragBF { unsigned int u[8]; v16bf v; };

__device__ __forceinline__ unsigned short f32_bf16(float f) {
    unsigned int u = __float_as_uint(f);
    u += 0x7FFFu + ((u >> 16) & 1u);          // round-to-nearest-even
    return (unsigned short)(u >> 16);
}
__device__ __forceinline__ float sigmoidf_(float x) {
    return 1.0f / (1.0f + __expf(-x));
}

#define NTHREADS 512   // 16 waves of 32

__global__ void f32_to_bf16_kernel(const float* __restrict__ in,
                                   unsigned short* __restrict__ out, long n) {
    long i = (long)blockIdx.x * blockDim.x + threadIdx.x;
    if (i < n) out[i] = f32_bf16(in[i]);
}

// One workgroup owns 16 batch rows for the whole T-step scan.
// LDS: Wt = [W;U]^T as [4U rows(n)][KP cols(k)] bf16 (resident, loaded once)
//      aT = staged [x_t ; h] tile, [16 rows][KP] bf16
//      zL = z scratch [16][4U] f32 ; cL = cell state [16][U] f32
template<int U, int FIN, bool WSEQ, bool WFIN>
__global__ __launch_bounds__(NTHREADS)
void lstm_scan_kernel(const unsigned short* __restrict__ xin, long sB, long sT,
                      const unsigned short* __restrict__ Wg,
                      const unsigned short* __restrict__ Ug,
                      const float* __restrict__ bias,
                      unsigned short* __restrict__ hseq,
                      float* __restrict__ hfin,
                      int T, int B) {
    constexpr int K   = FIN + U;
    constexpr int KP  = K + 8;            // even pad (bank stagger, dword loads)
    constexpr int G   = 4 * U;
    constexpr int NW  = NTHREADS / 32;
    constexpr int TPW = (G / 16) / NW;    // col-tiles per wave
    constexpr int K32 = K / 32;
    static_assert(TPW >= 1 && (G / 16) % NW == 0 && K % 32 == 0, "shape");

    extern __shared__ char smem[];
    unsigned short* Wt = (unsigned short*)smem;            // [G][KP]
    unsigned short* aT = Wt + (long)G * KP;                // [16][KP]
    float*          zL = (float*)(aT + 16 * KP);           // [16][G]
    float*          cL = zL + 16 * G;                      // [16][U]

    const int tid  = threadIdx.x;
    const int lane = tid & 31;
    const int wave = tid >> 5;
    const int b0   = blockIdx.x * 16;

    // Load transposed concatenated weights once: Wt[n][k] = cat(W,U)[k][n]
    for (int i = tid; i < G * K; i += NTHREADS) {
        int n = i / K, k = i % K;
        Wt[n * KP + k] = (k < FIN) ? Wg[(long)k * G + n]
                                   : Ug[(long)(k - FIN) * G + n];
    }
    for (int i = tid; i < 16 * KP; i += NTHREADS) aT[i] = 0;      // h(=0), x
    for (int i = tid; i < 16 * U;  i += NTHREADS) cL[i] = 0.0f;   // c = 0
    __syncthreads();

    for (int t = 0; t < T; ++t) {
        // ---- stage x_t tile (bf16) into aT[:, 0:FIN], dword-granular
        constexpr int DW = 16 * FIN / 2;
        for (int d = tid; d < DW; d += NTHREADS) {
            int row = d / (FIN / 2);
            int p   = d % (FIN / 2);
            const unsigned short* src =
                xin + (long)(b0 + row) * sB + (long)t * sT + 2 * p;
            *(unsigned int*)&aT[row * KP + 2 * p] = *(const unsigned int*)src;
            if (p == 0 && t + 1 < T)
                __builtin_prefetch(src + sT, 0, 1);   // global_prefetch_b8
        }
        __syncthreads();   // x staged + prev-step h writes visible

        // ---- z = [x_t ; h] @ cat(W,U)  via v_wmma_f32_16x16x32_bf16
        v8f acc[TPW];
        const v8f vzero = {0.f, 0.f, 0.f, 0.f, 0.f, 0.f, 0.f, 0.f};
        #pragma unroll
        for (int tw = 0; tw < TPW; ++tw) acc[tw] = vzero;

        for (int ks = 0; ks < K32; ++ks) {
            // A fragment (16x32 bf16): lanes 0-15 rows M, K=0..15;
            // lanes 16-31 same rows, K=16..31 (pairs per VGPR).
            FragBF a;
            const int m  = lane & 15;
            const int kA = ks * 32 + ((lane >> 4) << 4);
            const unsigned short* ar = &aT[m * KP + kA];
            #pragma unroll
            for (int j = 0; j < 4; ++j) {
                a.u[j]     = *(const unsigned int*)(ar + 2 * j);
                a.u[4 + j] = *(const unsigned int*)(ar + 8 + 2 * j);
            }
            #pragma unroll
            for (int tw = 0; tw < TPW; ++tw) {
                // B fragment (32x16 bf16): lane = column N, lanes 0-15 K=0..15,
                // lanes 16-31 K=16..31, K-pairs across 8 VGPRs (Wt is [N][K]).
                const int n  = (wave * TPW + tw) * 16 + (lane & 15);
                const int kB = ks * 32 + ((lane >> 4) << 4);
                const unsigned short* br = &Wt[n * KP + kB];
                FragBF bfr;
                #pragma unroll
                for (int j = 0; j < 8; ++j)
                    bfr.u[j] = *(const unsigned int*)(br + 2 * j);
                acc[tw] = __builtin_amdgcn_wmma_f32_16x16x32_bf16(
                    false, a.v, false, bfr.v, (short)0, acc[tw], false, false);
            }
        }
        // C layout: VGPR r -> M = r + 8*(lane>=16), N = lane&15
        #pragma unroll
        for (int tw = 0; tw < TPW; ++tw) {
            const int col   = (wave * TPW + tw) * 16 + (lane & 15);
            const int rbase = (lane >> 4) << 3;
            #pragma unroll
            for (int r = 0; r < 8; ++r)
                zL[(rbase + r) * G + col] = acc[tw][r];
        }
        __syncthreads();   // z visible

        // ---- gates: i,f = sigmoid; g = relu; c = f*c + i*g; h = o*relu(c)
        for (int i = tid; i < 16 * U; i += NTHREADS) {
            const int row = i / U, un = i % U;
            const float* zr = &zL[row * G];
            float zi = zr[un]         + bias[un];
            float zf = zr[U + un]     + bias[U + un];
            float zg = zr[2 * U + un] + bias[2 * U + un];
            float zo = zr[3 * U + un] + bias[3 * U + un];
            float c  = cL[row * U + un];
            c = sigmoidf_(zf) * c + sigmoidf_(zi) * fmaxf(zg, 0.0f);
            float h = sigmoidf_(zo) * fmaxf(c, 0.0f);
            cL[row * U + un] = c;
            unsigned short hb = f32_bf16(h);
            aT[row * KP + FIN + un] = hb;                 // feed next step's A
            if (WSEQ)
                hseq[((long)t * B + b0 + row) * U + un] = hb;
            if (WFIN && t == T - 1)
                hfin[(long)(b0 + row) * U + un] = h;
        }
        // loop-top __syncthreads orders gate writes vs next wmma reads
    }
}

__global__ void dense_head_kernel(const float* __restrict__ h3,
                                  const float* __restrict__ Wd1,
                                  const float* __restrict__ bd1,
                                  const float* __restrict__ Wd2,
                                  const float* __restrict__ bd2,
                                  float* __restrict__ out, int B) {
    int b = blockIdx.x * blockDim.x + threadIdx.x;
    if (b >= B) return;
    float hl[64];
    #pragma unroll
    for (int k = 0; k < 64; ++k) hl[k] = h3[b * 64 + k];
    float o = bd2[0];
    #pragma unroll
    for (int j = 0; j < 32; ++j) {
        float s = bd1[j];
        #pragma unroll
        for (int k = 0; k < 64; ++k) s += hl[k] * Wd1[k * 32 + j];
        o += fmaxf(s, 0.0f) * Wd2[j];
    }
    out[b] = o;
}

extern "C" void kernel_launch(void* const* d_in, const int* in_sizes, int n_in,
                              void* d_out, int out_size, void* d_ws, size_t ws_size,
                              hipStream_t stream) {
    (void)in_sizes; (void)n_in; (void)ws_size;
    const int B = 256, T = 512, F = 64, U1 = 128, U2 = 64, U3 = 64;

    const float* x   = (const float*)d_in[0];
    const float* W1  = (const float*)d_in[1];
    const float* Uw1 = (const float*)d_in[2];
    const float* b1  = (const float*)d_in[3];
    const float* W2  = (const float*)d_in[4];
    const float* Uw2 = (const float*)d_in[5];
    const float* b2  = (const float*)d_in[6];
    const float* W3  = (const float*)d_in[7];
    const float* Uw3 = (const float*)d_in[8];
    const float* b3  = (const float*)d_in[9];
    const float* Wd1 = (const float*)d_in[10];
    const float* bd1 = (const float*)d_in[11];
    const float* Wd2 = (const float*)d_in[12];
    const float* bd2 = (const float*)d_in[13];

    // workspace layout (256B aligned slices)
    char* ws = (char*)d_ws;
    size_t off = 0;
    auto carve = [&](size_t bytes) -> char* {
        char* p = ws + off;
        off += (bytes + 255) & ~(size_t)255;
        return p;
    };
    unsigned short* xbf = (unsigned short*)carve((size_t)B * T * F  * 2);
    unsigned short* h1s = (unsigned short*)carve((size_t)T * B * U1 * 2);
    unsigned short* h2s = (unsigned short*)carve((size_t)T * B * U2 * 2);
    float*          h3  = (float*)         carve((size_t)B * U3 * 4);
    unsigned short* w1b = (unsigned short*)carve((size_t)F  * 4 * U1 * 2);
    unsigned short* u1b = (unsigned short*)carve((size_t)U1 * 4 * U1 * 2);
    unsigned short* w2b = (unsigned short*)carve((size_t)U1 * 4 * U2 * 2);
    unsigned short* u2b = (unsigned short*)carve((size_t)U2 * 4 * U2 * 2);
    unsigned short* w3b = (unsigned short*)carve((size_t)U2 * 4 * U3 * 2);
    unsigned short* u3b = (unsigned short*)carve((size_t)U3 * 4 * U3 * 2);

    auto cvt = [&](const float* in, unsigned short* outp, long n) {
        f32_to_bf16_kernel<<<(unsigned)((n + 255) / 256), 256, 0, stream>>>(in, outp, n);
    };
    cvt(x,   xbf, (long)B * T * F);
    cvt(W1,  w1b, (long)F  * 4 * U1);
    cvt(Uw1, u1b, (long)U1 * 4 * U1);
    cvt(W2,  w2b, (long)U1 * 4 * U2);
    cvt(Uw2, u2b, (long)U2 * 4 * U2);
    cvt(W3,  w3b, (long)U2 * 4 * U3);
    cvt(Uw3, u3b, (long)U3 * 4 * U3);

    auto smemBytes = [](int Uu, int Fin) -> size_t {
        int Kk = Uu + Fin, KP = Kk + 8, G = 4 * Uu;
        return (size_t)(G * KP + 16 * KP) * 2 + (size_t)(16 * G + 16 * Uu) * 4;
    };
    const size_t s1 = smemBytes(U1, F);   // ~252 KB
    const size_t s2 = smemBytes(U2, U1);  // ~129 KB
    const size_t s3 = smemBytes(U3, U2);  // ~94 KB

    hipFuncSetAttribute(reinterpret_cast<const void*>(&lstm_scan_kernel<128, 64, true,  false>),
                        hipFuncAttributeMaxDynamicSharedMemorySize, (int)s1);
    hipFuncSetAttribute(reinterpret_cast<const void*>(&lstm_scan_kernel<64, 128, true,  false>),
                        hipFuncAttributeMaxDynamicSharedMemorySize, (int)s2);
    hipFuncSetAttribute(reinterpret_cast<const void*>(&lstm_scan_kernel<64,  64, false, true>),
                        hipFuncAttributeMaxDynamicSharedMemorySize, (int)s3);

    // Layer 1: input x is [B,T,F] -> element stride sB = T*F, sT = F
    lstm_scan_kernel<128, 64, true, false><<<B / 16, NTHREADS, s1, stream>>>(
        xbf, (long)T * F, (long)F, w1b, u1b, b1, h1s, nullptr, T, B);
    // Layer 2: input h1 is [T,B,U1] -> sB = U1, sT = B*U1
    lstm_scan_kernel<64, 128, true, false><<<B / 16, NTHREADS, s2, stream>>>(
        h1s, (long)U1, (long)B * U1, w2b, u2b, b2, h2s, nullptr, T, B);
    // Layer 3: input h2 is [T,B,U2]; only final h kept (f32)
    lstm_scan_kernel<64, 64, false, true><<<B / 16, NTHREADS, s3, stream>>>(
        h2s, (long)U2, (long)B * U2, w3b, u3b, b3, nullptr, h3, T, B);

    dense_head_kernel<<<1, 256, 0, stream>>>(h3, Wd1, bd1, Wd2, bd2,
                                             (float*)d_out, B);
}
